// attention_net_463856468022
// MI455X (gfx1250) — compile-verified, matrix operand order checked by reading, not run on hardware
//
#include <hip/hip_runtime.h>

typedef __attribute__((ext_vector_type(16))) _Float16 v16h;
typedef __attribute__((ext_vector_type(8)))  _Float16 v8h;
typedef __attribute__((ext_vector_type(8)))  float    v8f;

#define NANCH 1614

// ---------------------------------------------------------------------------
// f32 -> f16 convert with per-row K padding: dst[r*Kpad + k] = k<Ktot ? src : 0
// (rows*Kpad elements; Kpad % 64 == 0 so GEMM rows are 16B aligned, tail-safe)
// ---------------------------------------------------------------------------
__global__ void cvt_pad_kernel(const float* __restrict__ src, _Float16* __restrict__ dst,
                               int rows, int Ktot, int Kpad)
{
    int gid = blockIdx.x * 256 + threadIdx.x;
    if (gid >= rows * Kpad) return;
    int r = gid / Kpad, k = gid - r * Kpad;
    dst[gid] = (k < Ktot) ? (_Float16)src[(size_t)r * Ktot + k] : (_Float16)0.f;
}

// ---------------------------------------------------------------------------
// Implicit-GEMM conv (ks x ks, pad, stride) via WMMA, all-f16 operands.
// in : (B, Cin, H, W) f16; w: (Cout, Kpad) f16 (K-padded rows); bias f32
// out: (B, Cout, OH, OW) f32 ReLU; out_h: optional f16 mirror.
// grid: (ceil(OH*OW/16), Cout/128, B), block 256 (8 waves; wave -> 16 M rows)
// K-loop unrolled x2 (64 K per iteration, 2 WMMAs).
// ---------------------------------------------------------------------------
__global__ void conv_wmma(const _Float16* __restrict__ in, const _Float16* __restrict__ w,
                          const float* __restrict__ bias, float* __restrict__ out,
                          _Float16* __restrict__ out_h,
                          int Cin, int H, int W, int Cout, int OH, int OW,
                          int ks, int pad, int stride)
{
    const int b     = blockIdx.z;
    const int tid   = threadIdx.x;
    const int lane  = tid & 31;
    const int mBase = blockIdx.y * 128 + (tid >> 5) * 16;
    const int nBase = blockIdx.x * 16;
    const int N     = OH * OW;
    const int ks2   = ks * ks;
    const int Ktot  = Cin * ks2;
    const int Kpad  = (Ktot + 63) & ~63;

    // B tile stored [n][k] so lanes read contiguous 16B chunks (ds_load_b128)
    __shared__ _Float16 lds_b[2][16][32];

    const _Float16* inb = in + (size_t)b * Cin * H * W;
    v8f acc = {};

    const int m = mBase + (lane & 15);
    const _Float16* wrow = w + (size_t)m * Kpad;

    const int kloc = (lane & 16) ? 8 : 0;
    const int nn   = lane & 15;

    for (int k0 = 0; k0 < Kpad; k0 += 64) {
        // --- cooperative im2col: 2 x (32K x 16N) f16 tiles, 4 elems/thread ---
        #pragma unroll
        for (int t = 0; t < 4; ++t) {
            int idx  = tid * 4 + t;          // 0..1023
            int half = idx >> 9;             // 0/1
            int rem  = idx & 511;
            int kk   = rem >> 4;             // 0..31
            int nx   = rem & 15;             // 0..15
            int K    = k0 + half * 32 + kk;
            _Float16 v = (_Float16)0.f;
            int p = nBase + nx;
            if (K < Ktot && p < N) {
                int ic = K / ks2;
                int r  = K - ic * ks2;
                int kh = r / ks;
                int kw = r - kh * ks;
                int oh = p / OW, ow = p - oh * OW;
                int ih = oh * stride + kh - pad;
                int iw = ow * stride + kw - pad;
                if ((unsigned)ih < (unsigned)H && (unsigned)iw < (unsigned)W)
                    v = inb[((size_t)ic * H + ih) * W + iw];
            }
            lds_b[half][nx][kk] = v;
        }
        __syncthreads();

        #pragma unroll
        for (int h = 0; h < 2; ++h) {
            int kb = k0 + h * 32 + kloc;
            // A fragment: two contiguous 16B vector loads from padded f16 weights
            v8h alo = *(const v8h*)(wrow + kb);
            v8h ahi = *(const v8h*)(wrow + kb + 16);
            v16h af = __builtin_shufflevector(alo, ahi,
                0, 1, 2, 3, 4, 5, 6, 7, 8, 9, 10, 11, 12, 13, 14, 15);
            // B fragment: two contiguous 16B LDS vector loads
            v8h blo = *(const v8h*)&lds_b[h][nn][kloc];
            v8h bhi = *(const v8h*)&lds_b[h][nn][kloc + 16];
            v16h bf = __builtin_shufflevector(blo, bhi,
                0, 1, 2, 3, 4, 5, 6, 7, 8, 9, 10, 11, 12, 13, 14, 15);
            acc = __builtin_amdgcn_wmma_f32_16x16x32_f16(false, af, false, bf,
                                                         (short)0, acc, false, false);
        }
        __syncthreads();
    }

    // --- epilogue: bias + ReLU, f32 out (+ optional f16 mirror) ---
    int p = nBase + nn;
    if (p < N) {
        int oh = p / OW, ow = p - oh * OW;
        int mh = (lane & 16) ? 8 : 0;
        #pragma unroll
        for (int r = 0; r < 8; ++r) {
            int mm = mBase + mh + r;
            float v = acc[r] + bias[mm];
            v = v > 0.f ? v : 0.f;
            size_t oi = (((size_t)b * Cout + mm) * OH + oh) * OW + ow;
            out[oi] = v;
            if (out_h) out_h[oi] = (_Float16)v;
        }
    }
}

// ---------------------------------------------------------------------------
// FC via WMMA: C[M,N] = A[M,K] * W[N,K]^T + bias[N]. A,W f16, K % 64 == 0.
// grid: (ceil(N/16), ceil(M/16)), block 32. K-loop unrolled x2.
// ---------------------------------------------------------------------------
__global__ void fc_wmma(const _Float16* __restrict__ A, const _Float16* __restrict__ W,
                        const float* __restrict__ bias, float* __restrict__ C,
                        int M, int N, int K)
{
    int lane  = threadIdx.x & 31;
    int nBase = blockIdx.x * 16, mBase = blockIdx.y * 16;
    int mi = mBase + (lane & 15); if (mi >= M) mi = M - 1;
    int ni = nBase + (lane & 15); if (ni >= N) ni = N - 1;
    const _Float16* ar = A + (size_t)mi * K;
    const _Float16* wr = W + (size_t)ni * K;
    const int kloc = (lane & 16) ? 8 : 0;

    v8f acc = {};
    for (int k0 = 0; k0 < K; k0 += 64) {
        #pragma unroll
        for (int h = 0; h < 2; ++h) {
            int kb = k0 + h * 32 + kloc;
            v8h alo = *(const v8h*)(ar + kb);
            v8h ahi = *(const v8h*)(ar + kb + 16);
            v8h blo = *(const v8h*)(wr + kb);
            v8h bhi = *(const v8h*)(wr + kb + 16);
            v16h af = __builtin_shufflevector(alo, ahi,
                0, 1, 2, 3, 4, 5, 6, 7, 8, 9, 10, 11, 12, 13, 14, 15);
            v16h bf = __builtin_shufflevector(blo, bhi,
                0, 1, 2, 3, 4, 5, 6, 7, 8, 9, 10, 11, 12, 13, 14, 15);
            acc = __builtin_amdgcn_wmma_f32_16x16x32_f16(false, af, false, bf,
                                                         (short)0, acc, false, false);
        }
    }

    int nn  = lane & 15;
    int col = nBase + nn;
    if (col < N) {
        int mh = (lane & 16) ? 8 : 0;
        #pragma unroll
        for (int r = 0; r < 8; ++r) {
            int row = mBase + mh + r;
            if (row < M) C[(size_t)row * N + col] = acc[r] + bias[col];
        }
    }
}

// ---------------------------------------------------------------------------
// 1x1 tidy convs -> rpn_score (B, 1614)
// ---------------------------------------------------------------------------
__global__ void tidy_kernel(const float* __restrict__ d1, const float* __restrict__ d2,
                            const float* __restrict__ d3,
                            const float* t1w, const float* t1b,
                            const float* t2w, const float* t2b,
                            const float* t3w, const float* t3b,
                            float* __restrict__ rpn)
{
    int gid = blockIdx.x * 256 + threadIdx.x;
    if (gid >= 16 * NANCH) return;
    int b = gid / NANCH, j = gid - b * NANCH;
    float acc;
    if (j < 1176) {
        int c = j / 196, p = j - c * 196;
        const float* in = d1 + (size_t)b * 128 * 196 + p;
        acc = t1b[c];
        for (int ic = 0; ic < 128; ++ic) acc += in[ic * 196] * t1w[c * 128 + ic];
    } else if (j < 1470) {
        int jj = j - 1176;
        int c = jj / 49, p = jj - c * 49;
        const float* in = d2 + (size_t)b * 128 * 49 + p;
        acc = t2b[c];
        for (int ic = 0; ic < 128; ++ic) acc += in[ic * 49] * t2w[c * 128 + ic];
    } else {
        int jj = j - 1470;
        int c = jj / 16, p = jj - c * 16;
        const float* in = d3 + (size_t)b * 128 * 16 + p;
        acc = t3b[c];
        for (int ic = 0; ic < 128; ++ic) acc += in[ic * 16] * t3w[c * 128 + ic];
    }
    rpn[gid] = acc;
}

// ---------------------------------------------------------------------------
// Edge anchors (matches numpy: double math then int32 truncation, +PAD)
// ---------------------------------------------------------------------------
__device__ __forceinline__ void anchor_box(int i, float box[4])
{
    const double s13 = 1.2599210498948731906665444;  // 2^(1/3)
    const double s23 = 1.5874010519681993613971827;  // 2^(2/3)
    const double ars[3] = {0.667, 1.0, 1.5};
    int n; double stride, size; double scales[3];
    int idx = i;
    if (idx < 1176)      { n = 14; stride = 32.0;  size = 48.0;  scales[0]=s13; scales[1]=s23; scales[2]=0.0; }
    else if (idx < 1470) { idx -= 1176; n = 7; stride = 64.0;  size = 96.0;  scales[0]=s13; scales[1]=s23; scales[2]=0.0; }
    else                 { idx -= 1470; n = 4; stride = 128.0; size = 192.0; scales[0]=1.0; scales[1]=s13; scales[2]=s23; }
    int per = n * n;
    int blk = idx / per, j = idx - blk * per;
    double sc = scales[blk / 3], ar = ars[blk % 3];
    int iy = j / n, ix = j - iy * n;
    double cy = iy * stride + stride * 0.5;
    double cx = ix * stride + stride * 0.5;
    double h = size * sc / sqrt(ar);
    double w = size * sc * sqrt(ar);
    box[0] = (float)(int)(cy - h * 0.5 + 224.0);
    box[1] = (float)(int)(cx - w * 0.5 + 224.0);
    box[2] = (float)(int)(cy + h * 0.5 + 224.0);
    box[3] = (float)(int)(cx + w * 0.5 + 224.0);
}

// ---------------------------------------------------------------------------
// Hard NMS: 1 block (256 thr) per batch image; picks TOPN=4.
// ---------------------------------------------------------------------------
__global__ void nms_kernel(const float* __restrict__ rpn,
                           float* __restrict__ top_idx_out,
                           float* __restrict__ top_prob_out,
                           float* __restrict__ sel_boxes)
{
    int b   = blockIdx.x;
    int tid = threadIdx.x;
    const float* s = rpn + (size_t)b * NANCH;

    __shared__ unsigned char mask[NANCH];
    __shared__ float rv[256];
    __shared__ int   ri[256];
    __shared__ float chosen[4];
    __shared__ float chosenArea;
    __shared__ int   chosenIdx;

    for (int i = tid; i < NANCH; i += 256) mask[i] = 1;
    __syncthreads();

    for (int k = 0; k < 4; ++k) {
        float best = -3.0e38f;
        int   bi   = 1 << 30;
        for (int i = tid; i < NANCH; i += 256) {
            if (mask[i]) {
                float v = s[i];
                if (v > best) { best = v; bi = i; }
            }
        }
        rv[tid] = best; ri[tid] = bi;
        __syncthreads();
        for (int off = 128; off > 0; off >>= 1) {
            if (tid < off) {
                if (rv[tid + off] > rv[tid] ||
                    (rv[tid + off] == rv[tid] && ri[tid + off] < ri[tid])) {
                    rv[tid] = rv[tid + off]; ri[tid] = ri[tid + off];
                }
            }
            __syncthreads();
        }
        if (tid == 0) {
            int ci = ri[0];
            if (ci >= NANCH) ci = 0;       // degenerate all-masked case (argmax -> 0)
            chosenIdx = ci;
            float bx[4]; anchor_box(ci, bx);
            chosen[0] = bx[0]; chosen[1] = bx[1]; chosen[2] = bx[2]; chosen[3] = bx[3];
            chosenArea = (bx[2] - bx[0]) * (bx[3] - bx[1]);
            top_idx_out[b * 4 + k]  = (float)ci;
            top_prob_out[b * 4 + k] = s[ci];
            float* sb = sel_boxes + (b * 4 + k) * 4;
            sb[0] = bx[0]; sb[1] = bx[1]; sb[2] = bx[2]; sb[3] = bx[3];
        }
        __syncthreads();
        float by0 = chosen[0], bx0 = chosen[1], by1 = chosen[2], bx1 = chosen[3];
        float ba  = chosenArea;
        int   ci  = chosenIdx;
        for (int i = tid; i < NANCH; i += 256) {
            float bx[4]; anchor_box(i, bx);
            float area = (bx[2] - bx[0]) * (bx[3] - bx[1]);
            float yy0 = bx[0] > by0 ? bx[0] : by0;
            float xx0 = bx[1] > bx0 ? bx[1] : bx0;
            float yy1 = bx[2] < by1 ? bx[2] : by1;
            float xx1 = bx[3] < bx1 ? bx[3] : bx1;
            float ih = yy1 - yy0; if (ih < 0.f) ih = 0.f;
            float iw = xx1 - xx0; if (iw < 0.f) iw = 0.f;
            float inter = ih * iw;
            float iou = inter / (area + ba - inter);
            unsigned char keep = (iou < 0.25f) ? 1 : 0;
            mask[i] = mask[i] & keep;
            if (i == ci) mask[i] = 0;
        }
        __syncthreads();
    }
}

// ---------------------------------------------------------------------------
// Bilinear crop-resize of padded image -> (64, 3, 224, 224) f16
// ---------------------------------------------------------------------------
__global__ void crop_resize_kernel(const float* __restrict__ x,
                                   const float* __restrict__ sel_boxes,
                                   _Float16* __restrict__ pimg)
{
    long long gid = (long long)blockIdx.x * 256 + threadIdx.x;
    const long long total = 64LL * 3 * 224 * 224;
    if (gid >= total) return;
    int ox = (int)(gid % 224);
    long long t = gid / 224;
    int oy = (int)(t % 224); t /= 224;
    int c  = (int)(t % 3);
    int bp = (int)(t / 3);
    int b  = bp >> 2;

    const float* box = sel_boxes + bp * 4;
    float y0 = box[0], x0 = box[1], y1 = box[2], x1 = box[3];
    float ty = (float)oy / 223.f, tx = (float)ox / 223.f;
    float ys = y0 + ty * (y1 - y0 - 1.f);
    float xs = x0 + tx * (x1 - x0 - 1.f);
    int yi0 = (int)floorf(ys); yi0 = yi0 < 0 ? 0 : (yi0 > 895 ? 895 : yi0);
    int xi0 = (int)floorf(xs); xi0 = xi0 < 0 ? 0 : (xi0 > 895 ? 895 : xi0);
    int yi1 = yi0 + 1 > 895 ? 895 : yi0 + 1;
    int xi1 = xi0 + 1 > 895 ? 895 : xi0 + 1;
    float wy = ys - (float)yi0;
    float wx = xs - (float)xi0;

    const float* img = x + ((size_t)b * 3 + c) * 448 * 448;
    auto sample = [&](int yy, int xx) -> float {
        int yu = yy - 224, xu = xx - 224;
        if ((unsigned)yu < 448u && (unsigned)xu < 448u)
            return img[(size_t)yu * 448 + xu];
        return 0.f;
    };
    float v00 = sample(yi0, xi0), v01 = sample(yi0, xi1);
    float v10 = sample(yi1, xi0), v11 = sample(yi1, xi1);
    float r0 = v00 + (v10 - v00) * wy;   // y interp first (matches reference)
    float r1 = v01 + (v11 - v01) * wy;
    float outv = r0 + (r1 - r0) * wx;
    pimg[gid] = (_Float16)outv;
}

// ---------------------------------------------------------------------------
// mean over 7x7 spatial (already ReLU'd) -> pf16 (64, 2048) f16
// ---------------------------------------------------------------------------
__global__ void mean49_kernel(const float* __restrict__ fo, _Float16* __restrict__ pf)
{
    int gid = blockIdx.x * 256 + threadIdx.x;
    if (gid >= 64 * 2048) return;
    const float* p = fo + (size_t)gid * 49;
    float s = 0.f;
    #pragma unroll
    for (int i = 0; i < 49; ++i) s += p[i];
    pf[gid] = (_Float16)(s * (1.f / 49.f));
}

// ---------------------------------------------------------------------------
// concat_in f16 (16, 10240) = [pf16 reshaped (16,8192) | feature (16,2048)]
// ---------------------------------------------------------------------------
__global__ void concat_build_kernel(const _Float16* __restrict__ pf,
                                    const float* __restrict__ feature,
                                    _Float16* __restrict__ cin)
{
    int gid = blockIdx.x * 256 + threadIdx.x;
    if (gid >= 16 * 10240) return;
    int b = gid / 10240, j = gid - b * 10240;
    cin[gid] = (j < 8192) ? pf[(size_t)b * 8192 + j]
                          : (_Float16)feature[(size_t)b * 2048 + (j - 8192)];
}

__global__ void copy_kernel(const float* __restrict__ src, float* __restrict__ dst, int n)
{
    int gid = blockIdx.x * 256 + threadIdx.x;
    if (gid < n) dst[gid] = src[gid];
}

// ---------------------------------------------------------------------------
extern "C" void kernel_launch(void* const* d_in, const int* in_sizes, int n_in,
                              void* d_out, int out_size, void* d_ws, size_t ws_size,
                              hipStream_t stream)
{
    const float* x        = (const float*)d_in[0];
    const float* rpn_feat = (const float*)d_in[1];
    const float* feature  = (const float*)d_in[2];
    const float* resnet   = (const float*)d_in[3];
    const float* d1w = (const float*)d_in[4];  const float* d1b = (const float*)d_in[5];
    const float* d2w = (const float*)d_in[6];  const float* d2b = (const float*)d_in[7];
    const float* d3w = (const float*)d_in[8];  const float* d3b = (const float*)d_in[9];
    const float* t1w = (const float*)d_in[10]; const float* t1b = (const float*)d_in[11];
    const float* t2w = (const float*)d_in[12]; const float* t2b = (const float*)d_in[13];
    const float* t3w = (const float*)d_in[14]; const float* t3b = (const float*)d_in[15];
    const float* fw  = (const float*)d_in[16]; const float* fb  = (const float*)d_in[17];
    const float* pw  = (const float*)d_in[18]; const float* pb  = (const float*)d_in[19];
    const float* cw  = (const float*)d_in[20]; const float* cb  = (const float*)d_in[21];
    float* out = (float*)d_out;

    char* ws = (char*)d_ws;
    size_t off = 0;
    auto take = [&](size_t bytes) -> void* {
        void* p = ws + off;
        off = (off + bytes + 255) & ~(size_t)255;
        return p;
    };
    // f16 converted operands
    _Float16* rpnf16 = (_Float16*)take((size_t)16 * 2048 * 196 * 2);
    _Float16* d1w16  = (_Float16*)take((size_t)128 * 18432 * 2);
    _Float16* d2w16  = (_Float16*)take((size_t)128 * 1152 * 2);
    _Float16* d3w16  = (_Float16*)take((size_t)128 * 1152 * 2);
    _Float16* fw16   = (_Float16*)take((size_t)2048 * 192 * 2);   // K padded 147 -> 192
    _Float16* pw16   = (_Float16*)take((size_t)200 * 2048 * 2);
    _Float16* cw16   = (_Float16*)take((size_t)200 * 10240 * 2);
    // activations
    float*    d1   = (float*)   take((size_t)16 * 128 * 196 * 4);
    _Float16* d1h  = (_Float16*)take((size_t)16 * 128 * 196 * 2);
    float*    d2   = (float*)   take((size_t)16 * 128 * 49 * 4);
    _Float16* d2h  = (_Float16*)take((size_t)16 * 128 * 49 * 2);
    float*    d3   = (float*)   take((size_t)16 * 128 * 16 * 4);
    float*    rpn  = (float*)   take((size_t)16 * NANCH * 4);
    float*    selbox = (float*) take((size_t)16 * 4 * 4 * 4);
    _Float16* pimg = (_Float16*)take((size_t)64 * 3 * 224 * 224 * 2);
    float*    feato = (float*)  take((size_t)64 * 2048 * 49 * 4);
    _Float16* pf16 = (_Float16*)take((size_t)64 * 2048 * 2);
    _Float16* cin16 = (_Float16*)take((size_t)16 * 10240 * 2);

    auto cvt = [&](const float* s, _Float16* d, int rows, int Ktot, int Kpad) {
        int n = rows * Kpad;
        cvt_pad_kernel<<<(n + 255) / 256, 256, 0, stream>>>(s, d, rows, Ktot, Kpad);
    };
    // weights (K-padded rows) + input activations
    cvt(rpn_feat, rpnf16, 1, 16 * 2048 * 196, 16 * 2048 * 196);
    cvt(d1w, d1w16, 128, 18432, 18432);
    cvt(d2w, d2w16, 128, 1152, 1152);
    cvt(d3w, d3w16, 128, 1152, 1152);
    cvt(fw,  fw16, 2048, 147, 192);
    cvt(pw,  pw16, 1, 200 * 2048, 200 * 2048);
    cvt(cw,  cw16, 1, 200 * 10240, 200 * 10240);

    // proposal net convs (implicit GEMM + WMMA, f16 operands, f32 accum)
    conv_wmma<<<dim3(13, 1, 16), 256, 0, stream>>>(rpnf16, d1w16, d1b, d1, d1h,
        2048, 14, 14, 128, 14, 14, 3, 1, 1);
    conv_wmma<<<dim3(4, 1, 16), 256, 0, stream>>>(d1h, d2w16, d2b, d2, d2h,
        128, 14, 14, 128, 7, 7, 3, 1, 2);
    conv_wmma<<<dim3(1, 1, 16), 256, 0, stream>>>(d2h, d3w16, d3b, d3, (_Float16*)nullptr,
        128, 7, 7, 128, 4, 4, 3, 1, 2);

    // tidy 1x1 convs -> rpn_score
    tidy_kernel<<<(16 * NANCH + 255) / 256, 256, 0, stream>>>(
        d1, d2, d3, t1w, t1b, t2w, t2b, t3w, t3b, rpn);

    // hard NMS -> top_idx / top_n_prob / selected boxes
    nms_kernel<<<16, 256, 0, stream>>>(rpn, out + 19200, out + 19264, selbox);

    // crop + bilinear resize -> part images (f16)
    crop_resize_kernel<<<(int)((64LL * 3 * 224 * 224 + 255) / 256), 256, 0, stream>>>(
        x, selbox, pimg);

    // 7x7 s32 feature conv (WMMA) + spatial mean
    conv_wmma<<<dim3(4, 16, 64), 256, 0, stream>>>(pimg, fw16, fb, feato, (_Float16*)nullptr,
        3, 224, 224, 2048, 7, 7, 7, 3, 32);
    mean49_kernel<<<(64 * 2048 + 255) / 256, 256, 0, stream>>>(feato, pf16);

    // part classifier: (64,2048) @ (200,2048)^T -> part_logits
    fc_wmma<<<dim3(13, 4), 32, 0, stream>>>(pf16, pw16, pb, out + 6400, 64, 200, 2048);

    // concat features, concat classifier
    concat_build_kernel<<<(16 * 10240 + 255) / 256, 256, 0, stream>>>(pf16, feature, cin16);
    fc_wmma<<<dim3(13, 1), 32, 0, stream>>>(cin16, cw16, cb, out + 3200, 16, 200, 10240);

    // resnet_out passthrough
    copy_kernel<<<(3200 + 255) / 256, 256, 0, stream>>>(resnet, out, 3200);
}
